// QLlamaMLP_30820685316554
// MI455X (gfx1250) — 8-bit matrix operand order repaired and checked by reading, compiled, not run on hardware
//
#include <hip/hip_runtime.h>
#include <cstdint>
#include <cstddef>

typedef __attribute__((ext_vector_type(8))) int v8i;

#define EPSV  1e-6f
#define QMAXV 127.0f

__device__ __forceinline__ void async_copy_b128(void* lds_ptr, const void* gptr) {
  unsigned lds_off = (unsigned)(uintptr_t)lds_ptr;
  asm volatile("global_load_async_to_lds_b128 %0, %1, off"
               :
               : "v"(lds_off), "v"(gptr)
               : "memory");
}
__device__ __forceinline__ void wait_async0() {
  asm volatile("s_wait_asynccnt 0x0" ::: "memory");
}

__device__ __forceinline__ float warp_max32(float v) {
#pragma unroll
  for (int o = 16; o > 0; o >>= 1) v = fmaxf(v, __shfl_xor(v, o, 32));
  return v;
}
__device__ __forceinline__ float warp_sum32(float v) {
#pragma unroll
  for (int o = 16; o > 0; o >>= 1) v += __shfl_xor(v, o, 32);
  return v;
}
__device__ __forceinline__ float block_max256(float v, float* sm) {
  v = warp_max32(v);
  const int w = threadIdx.x >> 5, l = threadIdx.x & 31;
  if (l == 0) sm[w] = v;
  __syncthreads();
  float r = sm[0];
#pragma unroll
  for (int i = 1; i < 8; ++i) r = fmaxf(r, sm[i]);
  __syncthreads();
  return r;
}
__device__ __forceinline__ float block_sum256(float v, float* sm) {
  v = warp_sum32(v);
  const int w = threadIdx.x >> 5, l = threadIdx.x & 31;
  if (l == 0) sm[w] = v;
  __syncthreads();
  float r = 0.f;
#pragma unroll
  for (int i = 0; i < 8; ++i) r += sm[i];
  __syncthreads();
  return r;
}

__device__ __forceinline__ v8i load_frag_a(const signed char* sA, int rowBase, int lane) {
  const int m  = lane & 15;
  const int kh = lane >> 4;
  const signed char* p = sA + (rowBase + m) * 64 + kh * 8;
  v8i f;
  const int* q0 = (const int*)(p);
  const int* q1 = (const int*)(p + 16);
  const int* q2 = (const int*)(p + 32);
  const int* q3 = (const int*)(p + 48);
  f[0] = q0[0]; f[1] = q0[1];
  f[2] = q1[0]; f[3] = q1[1];
  f[4] = q2[0]; f[5] = q2[1];
  f[6] = q3[0]; f[7] = q3[1];
  return f;
}
__device__ __forceinline__ v8i load_frag_b(const signed char* sB, int colBase, int lane) {
  const int n  = lane & 15;
  const int kh = lane >> 4;
  const signed char* p = sB + (colBase + n) * 64 + kh * 8;
  const int2 x0 = *(const int2*)(p);
  const int2 x1 = *(const int2*)(p + 16);
  const int2 x2 = *(const int2*)(p + 32);
  const int2 x3 = *(const int2*)(p + 48);
  v8i f;
  f[0] = x0.x; f[1] = x0.y; f[2] = x1.x; f[3] = x1.y;
  f[4] = x2.x; f[5] = x2.y; f[6] = x3.x; f[7] = x3.y;
  return f;
}

__device__ __forceinline__ v8i wmma_iu8(v8i a, v8i b, v8i c) {
  return __builtin_amdgcn_wmma_i32_16x16x64_iu8(true, a, true, b, c, false, false);
}

__global__ __launch_bounds__(256) void quant_rows_kernel(
    const float* __restrict__ W, signed char* __restrict__ Wq,
    float* __restrict__ scales, int L) {
  __shared__ float sm[8];
  const long long row = blockIdx.x;
  const float* wr = W + row * (long long)L;
  float m = 0.f;
  for (int i = threadIdx.x; i < L; i += 256) m = fmaxf(m, fabsf(wr[i]));
  m = block_max256(m, sm);
  const float sc  = fmaxf(m / QMAXV, 1e-8f);
  const float inv = 1.f / sc;
  if (threadIdx.x == 0) scales[row] = sc;
  signed char* qr = Wq + row * (long long)L;
  for (int i = threadIdx.x; i < L; i += 256) {
    float v = rintf(wr[i] * inv);
    v = fminf(fmaxf(v, -QMAXV), QMAXV);
    qr[i] = (signed char)(int)v;
  }
}

__global__ __launch_bounds__(256) void rms_quant_kernel(
    const float* __restrict__ x, const float* __restrict__ g,
    signed char* __restrict__ q, float* __restrict__ s, int D) {
  __shared__ float sm[8];
  const long long t = blockIdx.x;
  const float* xr = x + t * (long long)D;
  float ss = 0.f;
  for (int i = threadIdx.x; i < D; i += 256) { const float v = xr[i]; ss += v * v; }
  const float tot  = block_sum256(ss, sm);
  const float rinv = rsqrtf(tot / (float)D + EPSV);
  float m = 0.f;
  for (int i = threadIdx.x; i < D; i += 256)
    m = fmaxf(m, fabsf(xr[i] * rinv * g[i]));
  m = block_max256(m, sm);
  const float sc  = fmaxf(m / QMAXV, 1e-8f);
  const float inv = 1.f / sc;
  if (threadIdx.x == 0) s[t] = sc;
  signed char* qr = q + t * (long long)D;
  for (int i = threadIdx.x; i < D; i += 256) {
    float v = rintf(xr[i] * rinv * g[i] * inv);
    v = fminf(fmaxf(v, -QMAXV), QMAXV);
    qr[i] = (signed char)(int)v;
  }
}

__global__ __launch_bounds__(256) void gemm_gateup_kernel(
    const signed char* __restrict__ qa, const float* __restrict__ sa,
    const signed char* __restrict__ Wgq, const float* __restrict__ wsg,
    const signed char* __restrict__ Wuq, const float* __restrict__ wsu,
    float* __restrict__ h, int D, long long I) {
  __shared__ __align__(16) signed char sA[2][128 * 64];
  __shared__ __align__(16) signed char sG[2][64 * 64];
  __shared__ __align__(16) signed char sU[2][64 * 64];

  const long long bm = (long long)blockIdx.x * 128;
  const long long bn = (long long)blockIdx.y * 64;
  const int tid  = threadIdx.x;
  const int lane = tid & 31;
  const int wave = tid >> 5;
  const int wm = wave & 1;
  const int wn = wave >> 1;

  v8i accg[4], accu[4];
#pragma unroll
  for (int i = 0; i < 4; ++i) {
    accg[i] = (v8i){0, 0, 0, 0, 0, 0, 0, 0};
    accu[i] = (v8i){0, 0, 0, 0, 0, 0, 0, 0};
  }

  const int ar = tid >> 1, as = (tid & 1) * 32;
  const int br = tid >> 2, bs = (tid & 3) * 16;
  const signed char* gA = qa  + (bm + ar) * (long long)D + as;
  const signed char* gG = Wgq + (bn + br) * (long long)D + bs;
  const signed char* gU = Wuq + (bn + br) * (long long)D + bs;

  auto stage = [&](int buf, int k0) {
    async_copy_b128(&sA[buf][ar * 64 + as],      gA + k0);
    async_copy_b128(&sA[buf][ar * 64 + as + 16], gA + k0 + 16);
    async_copy_b128(&sG[buf][br * 64 + bs],      gG + k0);
    async_copy_b128(&sU[buf][br * 64 + bs],      gU + k0);
  };

  stage(0, 0);
  wait_async0();
  __syncthreads();

  int cur = 0;
  for (int k0 = 0; k0 < D; k0 += 64) {
    if (k0 + 64 < D) stage(cur ^ 1, k0 + 64);

    const v8i bg = load_frag_b(sG[cur], wn * 16, lane);
    const v8i bu = load_frag_b(sU[cur], wn * 16, lane);
#pragma unroll
    for (int mt = 0; mt < 4; ++mt) {
      const v8i af = load_frag_a(sA[cur], wm * 64 + mt * 16, lane);
      accg[mt] = wmma_iu8(af, bg, accg[mt]);
      accu[mt] = wmma_iu8(af, bu, accu[mt]);
    }

    wait_async0();
    __syncthreads();
    cur ^= 1;
  }

  const int cn = lane & 15;
  const int ch = (lane >> 4) * 8;
  const long long gcol = bn + wn * 16 + cn;
  const float wg_s = wsg[gcol];
  const float wu_s = wsu[gcol];
#pragma unroll
  for (int mt = 0; mt < 4; ++mt) {
#pragma unroll
    for (int r = 0; r < 8; ++r) {
      const long long grow = bm + wm * 64 + mt * 16 + ch + r;
      const float sv = sa[grow];
      const float gf = (float)accg[mt][r] * sv * wg_s;
      const float uf = (float)accu[mt][r] * sv * wu_s;
      const float hv = (gf / (1.f + __expf(-gf))) * uf;
      h[grow * I + gcol] = hv;
    }
  }
}

__global__ __launch_bounds__(256) void gemm_down_kernel(
    const signed char* __restrict__ q2, const float* __restrict__ s2,
    const signed char* __restrict__ Wdq, const float* __restrict__ wsd,
    float* __restrict__ out, int I, int D) {
  __shared__ __align__(16) signed char sA[2][128 * 64];
  __shared__ __align__(16) signed char sB[2][128 * 64];

  const long long bm = (long long)blockIdx.x * 128;
  const long long bn = (long long)blockIdx.y * 128;
  const int tid  = threadIdx.x;
  const int lane = tid & 31;
  const int wave = tid >> 5;
  const int wm = wave & 1;
  const int wn = wave >> 1;

  v8i acc[4][2];
#pragma unroll
  for (int i = 0; i < 4; ++i)
#pragma unroll
    for (int j = 0; j < 2; ++j) acc[i][j] = (v8i){0, 0, 0, 0, 0, 0, 0, 0};

  const int ar = tid >> 1, as = (tid & 1) * 32;
  const signed char* gA = q2  + (bm + ar) * (long long)I + as;
  const signed char* gB = Wdq + (bn + ar) * (long long)I + as;

  auto stage = [&](int buf, int k0) {
    async_copy_b128(&sA[buf][ar * 64 + as],      gA + k0);
    async_copy_b128(&sA[buf][ar * 64 + as + 16], gA + k0 + 16);
    async_copy_b128(&sB[buf][ar * 64 + as],      gB + k0);
    async_copy_b128(&sB[buf][ar * 64 + as + 16], gB + k0 + 16);
  };

  stage(0, 0);
  wait_async0();
  __syncthreads();

  int cur = 0;
  for (int k0 = 0; k0 < I; k0 += 64) {
    if (k0 + 64 < I) stage(cur ^ 1, k0 + 64);

    v8i bf[2];
#pragma unroll
    for (int nt = 0; nt < 2; ++nt) bf[nt] = load_frag_b(sB[cur], wn * 32 + nt * 16, lane);
#pragma unroll
    for (int mt = 0; mt < 4; ++mt) {
      const v8i af = load_frag_a(sA[cur], wm * 64 + mt * 16, lane);
#pragma unroll
      for (int nt = 0; nt < 2; ++nt)
        acc[mt][nt] = wmma_iu8(af, bf[nt], acc[mt][nt]);
    }

    wait_async0();
    __syncthreads();
    cur ^= 1;
  }

  const int cn = lane & 15;
  const int ch = (lane >> 4) * 8;
#pragma unroll
  for (int nt = 0; nt < 2; ++nt) {
    const long long gcol = bn + wn * 32 + nt * 16 + cn;
    const float wd_s = wsd[gcol];
#pragma unroll
    for (int mt = 0; mt < 4; ++mt) {
#pragma unroll
      for (int r = 0; r < 8; ++r) {
        const long long grow = bm + wm * 64 + mt * 16 + ch + r;
        out[grow * (long long)D + gcol] = (float)acc[mt][nt][r] * s2[grow] * wd_s;
      }
    }
  }
}

extern "C" void kernel_launch(void* const* d_in, const int* in_sizes, int n_in,
                              void* d_out, int out_size, void* d_ws, size_t ws_size,
                              hipStream_t stream) {
  const float* x  = (const float*)d_in[0];
  const float* gw = (const float*)d_in[1];
  const float* Wg = (const float*)d_in[2];
  const float* Wu = (const float*)d_in[3];
  const float* Wd = (const float*)d_in[4];
  float* out = (float*)d_out;

  const long long D = in_sizes[1];
  const long long T = (long long)in_sizes[0] / D;
  const long long I = (long long)in_sizes[2] / D;

  char* p = (char*)d_ws;
  auto take = [&](size_t bytes) -> char* {
    char* r = p;
    p += (bytes + 255) & ~(size_t)255;
    return r;
  };
  signed char* qa  = (signed char*)take((size_t)(T * D));
  float*       sa  = (float*)      take((size_t)(T * 4));
  signed char* Wgq = (signed char*)take((size_t)(I * D));
  float*       wsg = (float*)      take((size_t)(I * 4));
  signed char* Wuq = (signed char*)take((size_t)(I * D));
  float*       wsu = (float*)      take((size_t)(I * 4));
  signed char* Wdq = (signed char*)take((size_t)(D * I));
  float*       wsd = (float*)      take((size_t)(D * 4));
  float*       h   = (float*)      take((size_t)(T * I * 4));
  signed char* q2  = (signed char*)take((size_t)(T * I));
  float*       s2  = (float*)      take((size_t)(T * 4));

  quant_rows_kernel<<<(unsigned)I, 256, 0, stream>>>(Wg, Wgq, wsg, (int)D);
  quant_rows_kernel<<<(unsigned)I, 256, 0, stream>>>(Wu, Wuq, wsu, (int)D);
  quant_rows_kernel<<<(unsigned)D, 256, 0, stream>>>(Wd, Wdq, wsd, (int)I);

  rms_quant_kernel<<<(unsigned)T, 256, 0, stream>>>(x, gw, qa, sa, (int)D);

  dim3 g1((unsigned)(T / 128), (unsigned)(I / 64));
  gemm_gateup_kernel<<<g1, 256, 0, stream>>>(qa, sa, Wgq, wsg, Wuq, wsu, h, (int)D, I);

  quant_rows_kernel<<<(unsigned)T, 256, 0, stream>>>(h, q2, s2, (int)I);

  dim3 g2((unsigned)(T / 128), (unsigned)(D / 128));
  gemm_down_kernel<<<g2, 256, 0, stream>>>(q2, s2, Wdq, wsd, out, (int)I, (int)D);
}
